// PerformerHead_446676599104
// MI455X (gfx1250) — compile-verified, matrix operand order checked by reading, not run on hardware
//
#include <hip/hip_runtime.h>
#include <hip/hip_bf16.h>
#include <math.h>

// ---------------- types & constants ----------------
typedef __bf16 v16bf __attribute__((ext_vector_type(16)));
typedef __bf16 v8bf  __attribute__((ext_vector_type(8)));
typedef float  v8f   __attribute__((ext_vector_type(8)));

static constexpr float RATIO = 0.06131392f;           // 266^-0.5
static constexpr float DN    = 0.35355339059327378f;  // 64^-0.25 (folded into proj)
// 0.5*dn*dn = 0.0625 used for diag terms

// ---------------- workspace byte offsets (bf16 weights stored transposed [N][K]) --
static constexpr size_t WQT_OFF  = 0;                             // [512][64]
static constexpr size_t WKT_OFF  = WQT_OFF + (size_t)512*64*2;
static constexpr size_t WVT_OFF  = WKT_OFF + (size_t)512*64*2;
static constexpr size_t WOT_OFF  = WVT_OFF + (size_t)512*64*2;    // [64][512]
static constexpr size_t PROJ_OFF = WOT_OFF + (size_t)64*512*2;    // [288][64] (scaled by DN, zero-padded)
static constexpr size_t W1T_OFF  = PROJ_OFF + (size_t)288*64*2;   // [256][64]
static constexpr size_t W2T_OFF  = W1T_OFF + (size_t)256*64*2;    // [64][256]
static constexpr size_t H_OFF    = W2T_OFF + (size_t)64*256*2;    // [256][128][64] bf16
static constexpr size_t BMAX_OFF = H_OFF + (size_t)256*128*64*2;  // [2048] f32
static constexpr size_t GMAX_OFF = BMAX_OFF + (size_t)2048*4;     // scalar
static constexpr size_t O_OFF    = GMAX_OFF + 64;                 // [256][128][512] bf16

// dynamic LDS sizes
static constexpr int D_LDS = (8192*3 + 288*64 + 128*32)*2 + 128*32*4 + (288 + 128*3)*4; // 113280
static constexpr int E_LDS = 128*64*4 + 128*64*2 + 128*128*2 + (64+128)*4;              // 82688

// ---------------- WMMA helpers ----------------
static __device__ __forceinline__ v8f wmma_bf16(v16bf a, v16bf b, v8f c) {
  return __builtin_amdgcn_wmma_f32_16x16x32_bf16(false, a, false, b, (short)0, c, false, false);
}

// Fragment whose "lane-held" dimension (A rows / B cols) is the leading storage
// dimension, K contiguous: per lane -> two 16-byte b128 loads.
// ISA 16-bit A 16x32 layout: lane half kh=0/8, runs K=[kh,kh+8) and [kh+16,kh+24).
static __device__ __forceinline__ v16bf ld_frag_vec(const __bf16* base, int r0, int ld,
                                                    int k0, int lane) {
  int rr = r0 + (lane & 15);
  int kh = (lane >> 4) * 8;
  const __bf16* p = base + (size_t)rr * ld + k0 + kh;
  v8bf lo = *(const v8bf*)p;
  v8bf hi = *(const v8bf*)(p + 16);
  return __builtin_shufflevector(lo, hi, 0, 1, 2, 3, 4, 5, 6, 7,
                                 8, 9, 10, 11, 12, 13, 14, 15);
}

// C fragment helper: lane holds col (lane&15), rows = rbase..rbase+7.
static __device__ __forceinline__ v8bf pack_bf8(v8f c) {
  v8bf r;
#pragma unroll
  for (int j = 0; j < 8; ++j) r[j] = (__bf16)c[j];
  return r;
}

// dst[128][64] = src[128][64] @ WT-slice + bias (WT stored [512][64], K-contig)
static __device__ __forceinline__ void proj_qkv(const __bf16* src, const __bf16* WT,
                                                const float* bias, int colbase,
                                                __bf16* dst, int wv, int lane) {
#pragma unroll
  for (int i = 0; i < 4; ++i) {
    int t = wv * 4 + i;
    int nt = t >> 2, dt = t & 3;
    float bb = bias[colbase + dt*16 + (lane & 15)];
    v8f c;
#pragma unroll
    for (int j = 0; j < 8; ++j) c[j] = bb;
#pragma unroll
    for (int kc = 0; kc < 2; ++kc) {
      v16bf a  = ld_frag_vec(src, nt*16, 64, kc*32, lane);
      v16bf bf = ld_frag_vec(WT, colbase + dt*16, 64, kc*32, lane);
      c = wmma_bf16(a, bf, c);
    }
    int col = dt*16 + (lane & 15);
    int rbase = nt*16 + ((lane >> 4) << 3);
#pragma unroll
    for (int j = 0; j < 8; ++j) dst[(rbase + j)*64 + col] = (__bf16)c[j];
  }
}

// transposed-output variant: dst_t[64][128] (contiguous per-lane 16B store)
static __device__ __forceinline__ void proj_qkv_t(const __bf16* src, const __bf16* WT,
                                                  const float* bias, int colbase,
                                                  __bf16* dst_t, int wv, int lane) {
#pragma unroll
  for (int i = 0; i < 4; ++i) {
    int t = wv * 4 + i;
    int nt = t >> 2, dt = t & 3;
    float bb = bias[colbase + dt*16 + (lane & 15)];
    v8f c;
#pragma unroll
    for (int j = 0; j < 8; ++j) c[j] = bb;
#pragma unroll
    for (int kc = 0; kc < 2; ++kc) {
      v16bf a  = ld_frag_vec(src, nt*16, 64, kc*32, lane);
      v16bf bf = ld_frag_vec(WT, colbase + dt*16, 64, kc*32, lane);
      c = wmma_bf16(a, bf, c);
    }
    int col = dt*16 + (lane & 15);
    int rbase = nt*16 + ((lane >> 4) << 3);
    *(v8bf*)(dst_t + (size_t)col * 128 + rbase) = pack_bf8(c);
  }
}

// chunkF[128][32] = mat[128][64] @ projs[ch*32:+32][64]^T  (row-major [n][m] store)
static __device__ __forceinline__ void dd_chunk(const __bf16* mat, const __bf16* projs,
                                                int ch, float* chunkF, int wv, int lane) {
#pragma unroll
  for (int i = 0; i < 2; ++i) {
    int t = wv * 2 + i;
    int nt = t >> 1, mt = t & 1;
    v8f c;
#pragma unroll
    for (int j = 0; j < 8; ++j) c[j] = 0.f;
#pragma unroll
    for (int kc = 0; kc < 2; ++kc) {
      v16bf a  = ld_frag_vec(mat, nt*16, 64, kc*32, lane);
      v16bf bf = ld_frag_vec(projs, ch*32 + mt*16, 64, kc*32, lane);
      c = wmma_bf16(a, bf, c);
    }
    int col = mt*16 + (lane & 15);
    int rbase = nt*16 + ((lane >> 4) << 3);
#pragma unroll
    for (int j = 0; j < 8; ++j) chunkF[(rbase + j)*32 + col] = c[j];
  }
}

// transposed store variant: outT[32][128] ([m][n]); per lane two float4 stores
static __device__ __forceinline__ void dd_chunk_t(const __bf16* mat, const __bf16* projs,
                                                  int ch, float* outT, int wv, int lane) {
#pragma unroll
  for (int i = 0; i < 2; ++i) {
    int t = wv * 2 + i;
    int nt = t >> 1, mt = t & 1;
    v8f c;
#pragma unroll
    for (int j = 0; j < 8; ++j) c[j] = 0.f;
#pragma unroll
    for (int kc = 0; kc < 2; ++kc) {
      v16bf a  = ld_frag_vec(mat, nt*16, 64, kc*32, lane);
      v16bf bf = ld_frag_vec(projs, ch*32 + mt*16, 64, kc*32, lane);
      c = wmma_bf16(a, bf, c);
    }
    int col = mt*16 + (lane & 15);            // m_local
    int rbase = nt*16 + ((lane >> 4) << 3);   // n
    *(float4*)(outT + (size_t)col * 128 + rbase)     = make_float4(c[0], c[1], c[2], c[3]);
    *(float4*)(outT + (size_t)col * 128 + rbase + 4) = make_float4(c[4], c[5], c[6], c[7]);
  }
}

// ---------------- kernel 0: weight conversion + transpose ----------------
__global__ __launch_bounds__(256) void perf_prep_kernel(
    const float* __restrict__ Wq, const float* __restrict__ Wk, const float* __restrict__ Wv,
    const float* __restrict__ Wo, const float* __restrict__ proj,
    const float* __restrict__ W1, const float* __restrict__ W2,
    __bf16* __restrict__ wqT, __bf16* __restrict__ wkT, __bf16* __restrict__ wvT,
    __bf16* __restrict__ woT, __bf16* __restrict__ projs,
    __bf16* __restrict__ w1T, __bf16* __restrict__ w2T) {
  int i = blockIdx.x * 256 + threadIdx.x;   // 32768 threads
  if (i < 32768) {
    int r = i >> 9, c = i & 511;            // Wq/Wk/Wv are [64][512]
    wqT[c*64 + r] = (__bf16)Wq[i];
    wkT[c*64 + r] = (__bf16)Wk[i];
    wvT[c*64 + r] = (__bf16)Wv[i];
    int k = i >> 6, n = i & 63;             // Wo is [512][64]
    woT[n*512 + k] = (__bf16)Wo[i];
  }
  if (i < 16384) {
    int k1 = i >> 8, n1 = i & 255;          // W1 is [64][256]
    w1T[n1*64 + k1] = (__bf16)W1[i];
    int k2 = i >> 6, n2 = i & 63;           // W2 is [256][64]
    w2T[n2*256 + k2] = (__bf16)W2[i];
  }
  if (i < 288*64) {
    projs[i] = (i < 266*64) ? (__bf16)(proj[i] * DN) : (__bf16)0.f;
  }
}

// ---------------- kernel 1: LN1 -> bf16 h ----------------
__global__ __launch_bounds__(128) void perf_ln1_kernel(
    const float* __restrict__ x, const float* __restrict__ g1, const float* __restrict__ b1,
    __bf16* __restrict__ h_ws) {
  int b = blockIdx.x, n = threadIdx.x;
  const float* row = x + ((size_t)b * 128 + n) * 64;
  float buf[64];
  float mu = 0.f;
#pragma unroll
  for (int g = 0; g < 16; ++g) {
    float4 f = *(const float4*)(row + g*4);
    buf[g*4+0] = f.x; buf[g*4+1] = f.y; buf[g*4+2] = f.z; buf[g*4+3] = f.w;
    mu += f.x + f.y + f.z + f.w;
  }
  mu *= (1.f / 64.f);
  float var = 0.f;
#pragma unroll
  for (int d = 0; d < 64; ++d) { float dv = buf[d] - mu; var += dv * dv; }
  var *= (1.f / 64.f);
  float inv = rsqrtf(var + 1e-5f);
  __bf16* dst = h_ws + ((size_t)b * 128 + n) * 64;
#pragma unroll
  for (int g = 0; g < 8; ++g) {
    v8bf o;
#pragma unroll
    for (int e = 0; e < 8; ++e)
      o[e] = (__bf16)((buf[g*8+e] - mu) * inv * g1[g*8+e] + b1[g*8+e]);
    *(v8bf*)(dst + g*8) = o;
  }
}

// ---------------- kernel 2: per-(b,h) key-feature max ----------------
__global__ __launch_bounds__(256) void perf_kmax_kernel(
    const __bf16* __restrict__ h_ws, const __bf16* __restrict__ WkT,
    const float* __restrict__ bk, const __bf16* __restrict__ projs,
    float* __restrict__ bmax) {
  __shared__ __bf16 h_s[128 * 64];
  __shared__ __bf16 k_s[128 * 64];
  __shared__ float red[8];
  int tid = threadIdx.x, lane = tid & 31, wv = tid >> 5;
  int bh = blockIdx.x, b = bh >> 3, hd = bh & 7;
  const uint4* hs4 = (const uint4*)(h_ws + (size_t)b * 8192);
  uint4* hd4 = (uint4*)h_s;
  for (int i = tid; i < 1024; i += 256) hd4[i] = hs4[i];
  __syncthreads();
  proj_qkv(h_s, WkT, bk, hd * 64, k_s, wv, lane);
  __syncthreads();
  float mx = -3.0e38f;
  for (int ch = 0; ch < 9; ++ch) {
#pragma unroll
    for (int i = 0; i < 2; ++i) {
      int t = wv * 2 + i;
      int nt = t >> 1, mt = t & 1;
      v8f c;
#pragma unroll
      for (int j = 0; j < 8; ++j) c[j] = 0.f;
#pragma unroll
      for (int kc = 0; kc < 2; ++kc) {
        v16bf a  = ld_frag_vec(k_s, nt*16, 64, kc*32, lane);
        v16bf bf = ld_frag_vec(projs, ch*32 + mt*16, 64, kc*32, lane);
        c = wmma_bf16(a, bf, c);
      }
      int m = ch*32 + mt*16 + (lane & 15);
      if (m < 266) {
#pragma unroll
        for (int j = 0; j < 8; ++j) mx = fmaxf(mx, c[j]);
      }
    }
  }
  for (int o = 16; o > 0; o >>= 1) mx = fmaxf(mx, __shfl_xor(mx, o, 32));
  if (lane == 0) red[wv] = mx;
  __syncthreads();
  if (tid == 0) {
    float m2 = red[0];
    for (int w = 1; w < 8; ++w) m2 = fmaxf(m2, red[w]);
    bmax[bh] = m2;
  }
}

// ---------------- kernel 3: global max reduce ----------------
__global__ __launch_bounds__(256) void perf_gmax_kernel(const float* __restrict__ bmax,
                                                        float* __restrict__ gmax) {
  __shared__ float red[256];
  int tid = threadIdx.x;
  float m = -3.0e38f;
  for (int i = tid; i < 2048; i += 256) m = fmaxf(m, bmax[i]);
  red[tid] = m;
  __syncthreads();
  for (int s = 128; s > 0; s >>= 1) {
    if (tid < s) red[tid] = fmaxf(red[tid], red[tid + s]);
    __syncthreads();
  }
  if (tid == 0) gmax[0] = red[0];
}

// ---------------- kernel 4: per-(b,h) FAVOR+ linear attention ----------------
__global__ __launch_bounds__(256) void perf_attn_kernel(
    const __bf16* __restrict__ h_ws,
    const __bf16* __restrict__ WqT, const float* __restrict__ bq,
    const __bf16* __restrict__ WkT, const float* __restrict__ bk,
    const __bf16* __restrict__ WvT, const float* __restrict__ bv,
    const __bf16* __restrict__ projs,
    const float* __restrict__ gmax_p,
    __bf16* __restrict__ o_ws) {
  extern __shared__ char smem[];
  __bf16* h_s    = (__bf16*)smem;            // [128][64]
  __bf16* kq_s   = h_s + 8192;               // [128][64] (k, later q, finally o staging)
  __bf16* v_t    = kq_s + 8192;              // [64][128]  (transposed v)
  __bf16* ctx_t  = v_t + 8192;               // [64][288]  (transposed ctx)
  __bf16* pk_s   = ctx_t + 64*288;           // kp phase: [32][128]; qp phase: [128][32]
  float*  chunkF = (float*)(pk_s + 128*32);  // kp phase: [32][128]; q phases: [128][32]
  float*  ksum_s = chunkF + 128*32;          // [288]
  float*  diag_s = ksum_s + 288;             // [128]
  float*  rmax_s = diag_s + 128;             // [128]
  float*  den_s  = rmax_s + 128;             // [128]

  const int tid = threadIdx.x, lane = tid & 31, wv = tid >> 5;
  const int bh = blockIdx.x, b = bh >> 3, hd = bh & 7;
  const float gmax = gmax_p[0];

  {
    const uint4* hs4 = (const uint4*)(h_ws + (size_t)b * 8192);
    uint4* hd4 = (uint4*)h_s;
    for (int i = tid; i < 1024; i += 256) hd4[i] = hs4[i];
  }
  if (tid < 128) { den_s[tid] = 0.f; rmax_s[tid] = -3.0e38f; }
  __syncthreads();

  // K (row-major) and V (transposed) projections
  proj_qkv(h_s, WkT, bk, hd * 64, kq_s, wv, lane);
  proj_qkv_t(h_s, WvT, bv, hd * 64, v_t, wv, lane);
  __syncthreads();

  if (tid < 128) {  // diag_k = 0.5*dn^2*sum(k^2)
    float s = 0.f;
#pragma unroll
    for (int g = 0; g < 8; ++g) {
      v8bf v = *(const v8bf*)(kq_s + tid*64 + g*8);
#pragma unroll
      for (int e = 0; e < 8; ++e) { float xx = (float)v[e]; s += xx * xx; }
    }
    diag_s[tid] = s * 0.0625f;
  }
  __syncthreads();

  // kp streamed in 32-wide feature chunks (transposed [m][n]) -> ksum, ctx = kp^T @ v
  for (int ch = 0; ch < 9; ++ch) {
    dd_chunk_t(kq_s, projs, ch, chunkF, wv, lane);   // chunkF as [32][128]
    __syncthreads();
    {   // exp: thread -> (m = tid>>3, 16 n's)
      int ml = tid >> 3;
      int n0 = (tid & 7) * 16;
      int mg = ch * 32 + ml;
      const float* src = chunkF + (size_t)ml * 128 + n0;
      float vals[16];
#pragma unroll
      for (int g = 0; g < 4; ++g) {
        float4 f = *(const float4*)(src + g*4);
        vals[g*4+0] = f.x; vals[g*4+1] = f.y; vals[g*4+2] = f.z; vals[g*4+3] = f.w;
      }
      v8bf o0, o1;
#pragma unroll
      for (int e = 0; e < 8; ++e) {
        float v0 = (mg < 266) ? RATIO * (__expf(vals[e]     - diag_s[n0 + e]     - gmax) + 1e-4f) : 0.f;
        float v1 = (mg < 266) ? RATIO * (__expf(vals[8 + e] - diag_s[n0 + 8 + e] - gmax) + 1e-4f) : 0.f;
        o0[e] = (__bf16)v0;
        o1[e] = (__bf16)v1;
      }
      *(v8bf*)(pk_s + (size_t)ml * 128 + n0)     = o0;   // pk as [32][128]
      *(v8bf*)(pk_s + (size_t)ml * 128 + n0 + 8) = o1;
    }
    __syncthreads();
    if (tid < 32) {   // ksum over rows of pkT (contiguous)
      float s = 0.f;
#pragma unroll
      for (int g = 0; g < 16; ++g) {
        v8bf v = *(const v8bf*)(pk_s + (size_t)tid * 128 + g*8);
#pragma unroll
        for (int e = 0; e < 8; ++e) s += (float)v[e];
      }
      ksum_s[ch*32 + tid] = s;
    }
    {   // ctx tile: 1 per wave, K=128, all fragments K-contiguous now
      int mt = wv >> 2, dt = wv & 3;
      v8f c;
#pragma unroll
      for (int j = 0; j < 8; ++j) c[j] = 0.f;
#pragma unroll
      for (int kc = 0; kc < 4; ++kc) {
        v16bf a  = ld_frag_vec(pk_s, mt*16, 128, kc*32, lane);  // kp^T [m][n]
        v16bf bf = ld_frag_vec(v_t, dt*16, 128, kc*32, lane);   // v^T  [d][n]
        c = wmma_bf16(a, bf, c);
      }
      int col = dt*16 + (lane & 15);                  // d
      int rbase = ch*32 + mt*16 + ((lane >> 4) << 3); // m
      *(v8bf*)(ctx_t + (size_t)col * 288 + rbase) = pack_bf8(c);
    }
    __syncthreads();
  }

  // Q projection (reuses kq_s), diag_q
  proj_qkv(h_s, WqT, bq, hd * 64, kq_s, wv, lane);
  __syncthreads();
  if (tid < 128) {
    float s = 0.f;
#pragma unroll
    for (int g = 0; g < 8; ++g) {
      v8bf v = *(const v8bf*)(kq_s + tid*64 + g*8);
#pragma unroll
      for (int e = 0; e < 8; ++e) { float xx = (float)v[e]; s += xx * xx; }
    }
    diag_s[tid] = s * 0.0625f;
  }
  __syncthreads();

  // pass 1: per-row max of dd_q (chunkF as [n][32])
  for (int ch = 0; ch < 9; ++ch) {
    dd_chunk(kq_s, projs, ch, chunkF, wv, lane);
    __syncthreads();
    if (tid < 128) {
      float m = rmax_s[tid];
      const float* src = chunkF + (size_t)tid * 32;
#pragma unroll
      for (int g = 0; g < 8; ++g) {
        float4 f = *(const float4*)(src + g*4);
        int mb = ch*32 + g*4;
        if (mb + 0 < 266) m = fmaxf(m, f.x);
        if (mb + 1 < 266) m = fmaxf(m, f.y);
        if (mb + 2 < 266) m = fmaxf(m, f.z);
        if (mb + 3 < 266) m = fmaxf(m, f.w);
      }
      rmax_s[tid] = m;
    }
    __syncthreads();
  }

  // pass 2: qp chunks -> denominator + o = qp @ ctx (WMMA accumulate)
  v8f oacc[4];
#pragma unroll
  for (int i = 0; i < 4; ++i)
#pragma unroll
    for (int j = 0; j < 8; ++j) oacc[i][j] = 0.f;

  for (int ch = 0; ch < 9; ++ch) {
    dd_chunk(kq_s, projs, ch, chunkF, wv, lane);
    __syncthreads();
    if (tid < 128) {
      const float* src = chunkF + (size_t)tid * 32;
      float diag = diag_s[tid], rm = rmax_s[tid];
      float vals[32];
#pragma unroll
      for (int g = 0; g < 8; ++g) {
        float4 f = *(const float4*)(src + g*4);
        vals[g*4+0] = f.x; vals[g*4+1] = f.y; vals[g*4+2] = f.z; vals[g*4+3] = f.w;
      }
      float dsum = 0.f;
#pragma unroll
      for (int j = 0; j < 32; ++j) {
        int m = ch * 32 + j;
        float val = 0.f;
        if (m < 266) val = RATIO * (__expf(vals[j] - diag - rm) + 1e-4f);
        vals[j] = val;
        dsum += val * ksum_s[m];
      }
      den_s[tid] += dsum;
#pragma unroll
      for (int g = 0; g < 4; ++g) {
        v8bf o;
#pragma unroll
        for (int e = 0; e < 8; ++e) o[e] = (__bf16)vals[g*8+e];
        *(v8bf*)(pk_s + (size_t)tid * 32 + g*8) = o;   // pk as [128][32]
      }
    }
    __syncthreads();
#pragma unroll
    for (int i = 0; i < 4; ++i) {
      int t = wv * 4 + i;
      int nt = t >> 2, dt = t & 3;
      v16bf a  = ld_frag_vec(pk_s, nt*16, 32, 0, lane);
      v16bf bf = ld_frag_vec(ctx_t, dt*16, 288, ch*32, lane);
      oacc[i] = wmma_bf16(a, bf, oacc[i]);
    }
    __syncthreads();
  }

  // stage o into kq_s (dead), then coalesced b128 global stores
#pragma unroll
  for (int i = 0; i < 4; ++i) {
    int t = wv * 4 + i;
    int nt = t >> 2, dt = t & 3;
    int col = dt*16 + (lane & 15);
    int rbase = nt*16 + ((lane >> 4) << 3);
#pragma unroll
    for (int j = 0; j < 8; ++j) {
      int n = rbase + j;
      kq_s[n*64 + col] = (__bf16)(oacc[i][j] / den_s[n]);
    }
  }
  __syncthreads();
  {
    __bf16* odst = o_ws + (size_t)b * (128 * 512) + hd * 64;
    const uint4* src4 = (const uint4*)kq_s;   // [128][64] bf16 = 8 uint4 per row
    for (int i = tid; i < 1024; i += 256) {
      int r = i >> 3, c8 = i & 7;
      *(uint4*)(odst + (size_t)r * 512 + c8 * 8) = src4[i];
    }
  }
}

// ---------------- kernel 5: out-proj + FF + head ----------------
__global__ __launch_bounds__(256) void perf_tail_kernel(
    const float* __restrict__ x, const __bf16* __restrict__ o_ws,
    const __bf16* __restrict__ WoT, const float* __restrict__ bo,
    const float* __restrict__ g2, const float* __restrict__ b2,
    const __bf16* __restrict__ W1T, const float* __restrict__ c1,
    const __bf16* __restrict__ W2T, const float* __restrict__ c2,
    const float* __restrict__ g3, const float* __restrict__ b3,
    const float* __restrict__ Wf1, const float* __restrict__ bf1,
    const float* __restrict__ Wf2, const float* __restrict__ bf2,
    float* __restrict__ out) {
  extern __shared__ char smem[];
  float*  t_s    = (float*)smem;             // [128][64] f32
  __bf16* h2_s   = (__bf16*)(t_s + 128*64);  // [128][64] bf16
  __bf16* scr    = h2_s + 8192;              // [128][128] bf16 (also [128][64] f32)
  float*  pooled = (float*)(scr + 16384);    // [64]
  float*  hbuf   = pooled + 64;              // [128]

  const int tid = threadIdx.x, lane = tid & 31, wv = tid >> 5;
  const int b = blockIdx.x;
  const float* xsrc = x + (size_t)b * 8192;
  {
    const float4* xs4 = (const float4*)xsrc;
    float4* td4 = (float4*)t_s;
    for (int i = tid; i < 2048; i += 256) td4[i] = xs4[i];
  }
  __syncthreads();

  // attn output projection: t += o_all @ Wo + bo   (K = 512, staged 128 at a time)
  v8f acc[4];
#pragma unroll
  for (int i = 0; i < 4; ++i) {
    int dt = (wv * 4 + i) & 3;
    float bb = bo[dt*16 + (lane & 15)];
#pragma unroll
    for (int j = 0; j < 8; ++j) acc[i][j] = bb;
  }
  const __bf16* osrc = o_ws + (size_t)b * (128 * 512);
  for (int kc = 0; kc < 4; ++kc) {
    for (int i = tid; i < 1024; i += 256) {    // [128][64] bf16 tile, 16B chunks
      int r = i >> 3, c16 = i & 7;
      ((uint4*)scr)[i] = *(const uint4*)(osrc + (size_t)r * 512 + kc*128 + c16*8);
    }
    __syncthreads();
#pragma unroll
    for (int i = 0; i < 4; ++i) {
      int t = wv * 4 + i;
      int nt = t >> 2, dt = t & 3;
#pragma unroll
      for (int s = 0; s < 4; ++s) {
        v16bf a  = ld_frag_vec(scr, nt*16, 128, s*32, lane);
        v16bf bf = ld_frag_vec(WoT, dt*16, 512, kc*128 + s*32, lane);
        acc[i] = wmma_bf16(a, bf, acc[i]);
      }
    }
    __syncthreads();
  }
#pragma unroll
  for (int i = 0; i < 4; ++i) {
    int t = wv * 4 + i;
    int nt = t >> 2, dt = t & 3;
    int col = dt*16 + (lane & 15);
    int rbase = nt*16 + ((lane >> 4) << 3);
#pragma unroll
    for (int j = 0; j < 8; ++j) t_s[(rbase + j)*64 + col] += acc[i][j];
  }
  __syncthreads();

  // LN2 -> h2 bf16
  if (tid < 128) {
    const float* r = t_s + (size_t)tid * 64;
    float buf[64];
    float mu = 0.f;
#pragma unroll
    for (int g = 0; g < 16; ++g) {
      float4 f = *(const float4*)(r + g*4);
      buf[g*4+0] = f.x; buf[g*4+1] = f.y; buf[g*4+2] = f.z; buf[g*4+3] = f.w;
      mu += f.x + f.y + f.z + f.w;
    }
    mu *= (1.f / 64.f);
    float var = 0.f;
#pragma unroll
    for (int d = 0; d < 64; ++d) { float dv = buf[d] - mu; var += dv * dv; }
    var *= (1.f / 64.f);
    float inv = rsqrtf(var + 1e-5f);
#pragma unroll
    for (int g = 0; g < 8; ++g) {
      v8bf o;
#pragma unroll
      for (int e = 0; e < 8; ++e)
        o[e] = (__bf16)((buf[g*8+e] - mu) * inv * g2[g*8+e] + b2[g*8+e]);
      *(v8bf*)(h2_s + (size_t)tid * 64 + g*8) = o;
    }
  }
  __syncthreads();

  // FF: t += gelu(h2@W1 + c1) @ W2 + c2   (hidden chunked by 64)
  v8f facc[4];
#pragma unroll
  for (int i = 0; i < 4; ++i) {
    int dt = (wv * 4 + i) & 3;
    float bb = c2[dt*16 + (lane & 15)];
#pragma unroll
    for (int j = 0; j < 8; ++j) facc[i][j] = bb;
  }
  for (int hc = 0; hc < 4; ++hc) {
#pragma unroll
    for (int i = 0; i < 4; ++i) {   // g chunk [128][64] with exact GELU
      int t = wv * 4 + i;
      int nt = t >> 2, ct = t & 3;
      float bb = c1[hc*64 + ct*16 + (lane & 15)];
      v8f c;
#pragma unroll
      for (int j = 0; j < 8; ++j) c[j] = bb;
#pragma unroll
      for (int kc = 0; kc < 2; ++kc) {
        v16bf a  = ld_frag_vec(h2_s, nt*16, 64, kc*32, lane);
        v16bf bf = ld_frag_vec(W1T, hc*64 + ct*16, 64, kc*32, lane);
        c = wmma_bf16(a, bf, c);
      }
      int col = ct*16 + (lane & 15);
      int rbase = nt*16 + ((lane >> 4) << 3);
#pragma unroll
      for (int j = 0; j < 8; ++j) {
        float xg = c[j];
        float gl = 0.5f * xg * (1.f + erff(xg * 0.70710678f));
        scr[(rbase + j)*64 + col] = (__bf16)gl;
      }
    }
    __syncthreads();
#pragma unroll
    for (int i = 0; i < 4; ++i) {
      int t = wv * 4 + i;
      int nt = t >> 2, dt = t & 3;
#pragma unroll
      for (int kc = 0; kc < 2; ++kc) {
        v16bf a  = ld_frag_vec(scr, nt*16, 64, kc*32, lane);
        v16bf bf = ld_frag_vec(W2T, dt*16, 256, hc*64 + kc*32, lane);
        facc[i] = wmma_bf16(a, bf, facc[i]);
      }
    }
    __syncthreads();
  }
#pragma unroll
  for (int i = 0; i < 4; ++i) {
    int t = wv * 4 + i;
    int nt = t >> 2, dt = t & 3;
    int col = dt*16 + (lane & 15);
    int rbase = nt*16 + ((lane >> 4) << 3);
#pragma unroll
    for (int j = 0; j < 8; ++j) t_s[(rbase + j)*64 + col] += facc[i][j];
  }
  __syncthreads();

  // LN3(t + tokens) -> f32 rows in scr
  float* outf = (float*)scr;
  if (tid < 128) {
    const float* r = t_s + (size_t)tid * 64;
    const float* xr = xsrc + (size_t)tid * 64;
    float buf[64];
    float mu = 0.f;
#pragma unroll
    for (int g = 0; g < 16; ++g) {
      float4 f = *(const float4*)(r + g*4);
      float4 xf = *(const float4*)(xr + g*4);
      buf[g*4+0] = f.x + xf.x; buf[g*4+1] = f.y + xf.y;
      buf[g*4+2] = f.z + xf.z; buf[g*4+3] = f.w + xf.w;
      mu += buf[g*4+0] + buf[g*4+1] + buf[g*4+2] + buf[g*4+3];
    }
    mu *= (1.f / 64.f);
    float var = 0.f;
#pragma unroll
    for (int d = 0; d < 64; ++d) { float dv = buf[d] - mu; var += dv * dv; }
    var *= (1.f / 64.f);
    float inv = rsqrtf(var + 1e-5f);
#pragma unroll
    for (int g = 0; g < 16; ++g) {
      float4 o;
      o.x = (buf[g*4+0] - mu) * inv * g3[g*4+0] + b3[g*4+0];
      o.y = (buf[g*4+1] - mu) * inv * g3[g*4+1] + b3[g*4+1];
      o.z = (buf[g*4+2] - mu) * inv * g3[g*4+2] + b3[g*4+2];
      o.w = (buf[g*4+3] - mu) * inv * g3[g*4+3] + b3[g*4+3];
      *(float4*)(outf + (size_t)tid * 64 + g*4) = o;
    }
  }
  __syncthreads();

  if (tid < 64) {  // mean pool over tokens
    float s = 0.f;
    for (int n = 0; n < 128; ++n) s += outf[n*64 + tid];
    pooled[tid] = s * (1.f / 128.f);
  }
  __syncthreads();

  if (tid < 128) { // relu(pooled @ Wf1 + bf1)
    float s = bf1[tid];
    for (int d = 0; d < 64; ++d) s += pooled[d] * Wf1[d*128 + tid];
    hbuf[tid] = fmaxf(s, 0.f);
  }
  __syncthreads();

  if (tid == 0) {
    float s = bf2[0];
    for (int j = 0; j < 128; ++j) s += hbuf[j] * Wf2[j];
    out[b] = s;
  }
}

// ---------------- launch ----------------
extern "C" void kernel_launch(void* const* d_in, const int* in_sizes, int n_in,
                              void* d_out, int out_size, void* d_ws, size_t ws_size,
                              hipStream_t stream) {
  (void)in_sizes; (void)n_in; (void)out_size; (void)ws_size;
  const float* x    = (const float*)d_in[0];
  const float* g1   = (const float*)d_in[1];
  const float* b1   = (const float*)d_in[2];
  const float* Wq   = (const float*)d_in[3];
  const float* bq   = (const float*)d_in[4];
  const float* Wk   = (const float*)d_in[5];
  const float* bk   = (const float*)d_in[6];
  const float* Wv   = (const float*)d_in[7];
  const float* bv   = (const float*)d_in[8];
  const float* Wo   = (const float*)d_in[9];
  const float* bo   = (const float*)d_in[10];
  const float* proj = (const float*)d_in[11];
  const float* g2   = (const float*)d_in[12];
  const float* b2   = (const float*)d_in[13];
  const float* W1   = (const float*)d_in[14];
  const float* c1   = (const float*)d_in[15];
  const float* W2   = (const float*)d_in[16];
  const float* c2   = (const float*)d_in[17];
  const float* g3   = (const float*)d_in[18];
  const float* b3   = (const float*)d_in[19];
  const float* Wf1  = (const float*)d_in[20];
  const float* bf1  = (const float*)d_in[21];
  const float* Wf2  = (const float*)d_in[22];
  const float* bf2  = (const float*)d_in[23];
  float* out = (float*)d_out;

  char* ws = (char*)d_ws;
  __bf16* wqT   = (__bf16*)(ws + WQT_OFF);
  __bf16* wkT   = (__bf16*)(ws + WKT_OFF);
  __bf16* wvT   = (__bf16*)(ws + WVT_OFF);
  __bf16* woT   = (__bf16*)(ws + WOT_OFF);
  __bf16* projs = (__bf16*)(ws + PROJ_OFF);
  __bf16* w1T   = (__bf16*)(ws + W1T_OFF);
  __bf16* w2T   = (__bf16*)(ws + W2T_OFF);
  __bf16* h_ws  = (__bf16*)(ws + H_OFF);
  float*  bmax  = (float*)(ws + BMAX_OFF);
  float*  gmaxp = (float*)(ws + GMAX_OFF);
  __bf16* o_ws  = (__bf16*)(ws + O_OFF);

  (void)hipFuncSetAttribute(reinterpret_cast<const void*>(perf_attn_kernel),
                            hipFuncAttributeMaxDynamicSharedMemorySize, D_LDS);
  (void)hipFuncSetAttribute(reinterpret_cast<const void*>(perf_tail_kernel),
                            hipFuncAttributeMaxDynamicSharedMemorySize, E_LDS);

  perf_prep_kernel<<<128, 256, 0, stream>>>(Wq, Wk, Wv, Wo, proj, W1, W2,
                                            wqT, wkT, wvT, woT, projs, w1T, w2T);
  perf_ln1_kernel<<<256, 128, 0, stream>>>(x, g1, b1, h_ws);
  perf_kmax_kernel<<<2048, 256, 0, stream>>>(h_ws, wkT, bk, projs, bmax);
  perf_gmax_kernel<<<1, 256, 0, stream>>>(bmax, gmaxp);
  perf_attn_kernel<<<2048, 256, D_LDS, stream>>>(h_ws, wqT, bq, wkT, bk, wvT, bv,
                                                 projs, gmaxp, o_ws);
  perf_tail_kernel<<<256, 256, E_LDS, stream>>>(x, o_ws, woT, bo, g2, b2, w1T, c1,
                                                w2T, c2, g3, b3, Wf1, bf1, Wf2, bf2, out);
}
